// SHR_ReLABlock_27900107555069
// MI455X (gfx1250) — matrix operand order repaired and checked by reading, not compiled, not run on hardware
//
#include <hip/hip_runtime.h>
#include <hip/hip_bf16.h>
#include <math.h>

// ---------------------------------------------------------------------------
// Problem constants (match reference)
// ---------------------------------------------------------------------------
#define BB 2
#define NN 2048
#define DIM 256
#define HH 8
#define DH 64
#define INNER 512            // H*DH
#define HID 1024
#define D3 768               // 3*DIM
#define ROWS (BB * NN)       // 4096 token rows

typedef __bf16 bf16;
typedef __attribute__((ext_vector_type(16))) __bf16 v16bf;
typedef __attribute__((ext_vector_type(8)))  float  v8f;

union Frag16 { v16bf v; unsigned int u[8]; };

// ---- CDNA5 async global->LDS copy (no VGPR round-trip), guarded -----------
#if __has_builtin(__builtin_amdgcn_global_load_async_to_lds_b128) && \
    __has_builtin(__builtin_amdgcn_s_wait_asynccnt)
#define USE_ASYNC_LDS 1
typedef int v4i_ __attribute__((ext_vector_type(4)));
typedef __attribute__((address_space(1))) v4i_ gv4i;   // global (AS1)
typedef __attribute__((address_space(3))) v4i_ lv4i;   // LDS (AS3)
static __device__ __forceinline__ void async_copy16(const void* g, void* l) {
  __builtin_amdgcn_global_load_async_to_lds_b128((gv4i*)g, (lv4i*)l, 0, 0);
}
static __device__ __forceinline__ void async_wait0() {
  __builtin_amdgcn_s_wait_asynccnt(0);
}
#else
#define USE_ASYNC_LDS 0
static __device__ __forceinline__ void async_copy16(const void* g, void* l) {
  *(uint4*)l = *(const uint4*)g;
}
static __device__ __forceinline__ void async_wait0() {}
#endif

static __device__ __forceinline__ v8f wmma_bf16(v16bf a, v16bf b, v8f c) {
  // D = A(16x32 bf16) * B(32x16 bf16) + C(16x16 f32)
  return __builtin_amdgcn_wmma_f32_16x16x32_bf16(
      /*neg_a=*/false, a, /*neg_b=*/false, b,
      /*c_mod=*/(short)0, c, /*reuse_a=*/false, /*reuse_b=*/false);
}

// A-matrix 16x32 bf16 per-lane K pattern (contiguous, matches B): VGPR i, lane-half h:
// kbase = h*16 + i*2  (pair kbase, kbase+1 packed in a dword)
static __device__ __forceinline__ int a_kbase(int i, int half) {
  return half * 16 + i * 2;
}

static __device__ __forceinline__ unsigned short f2bf_bits(float f) {
  bf16 h = (bf16)f;
  unsigned short u;
  __builtin_memcpy(&u, &h, 2);
  return u;
}

// ---------------------------------------------------------------------------
// Pack fp32 weight W[K x Nc] (row-major) into B-fragment-linear bf16 dwords.
// dword index j = ((kt * (Nc/16) + ntile) * 32 + lane) * 8 + i
//   half = lane>>4, nl = lane&15; kk = kt*32 + half*16 + 2i; n = ntile*16+nl
//   value = bf16(W[kk][n]) | bf16(W[kk+1][n]) << 16
// Consumer: each lane reads 8 consecutive dwords -> 2 coalesced b128 loads.
// ---------------------------------------------------------------------------
__global__ void pack_w_kernel(const float* __restrict__ W, unsigned int* __restrict__ Wp,
                              int K, int Nc) {
  int j = blockIdx.x * 256 + threadIdx.x;
  int total = (K * Nc) >> 1;
  if (j >= total) return;
  int i    = j & 7;
  int lane = (j >> 3) & 31;
  int tile = j >> 8;
  int tilesN = Nc >> 4;
  int ntile = tile % tilesN;
  int kt    = tile / tilesN;
  int half = lane >> 4;
  int nl   = lane & 15;
  int kk = kt * 32 + half * 16 + 2 * i;
  int n  = ntile * 16 + nl;
  unsigned int lo = f2bf_bits(W[(size_t)kk * Nc + n]);
  unsigned int hi = f2bf_bits(W[(size_t)(kk + 1) * Nc + n]);
  Wp[j] = lo | (hi << 16);
}

// ---------------------------------------------------------------------------
// Block-wide reduction helper (256 threads)
// ---------------------------------------------------------------------------
static __device__ __forceinline__ float block_sum(float v, float* red) {
  int t = threadIdx.x;
  red[t] = v;
  __syncthreads();
  for (int o = 128; o > 0; o >>= 1) {
    if (t < o) red[t] += red[t + o];
    __syncthreads();
  }
  float r = red[0];
  __syncthreads();
  return r;
}

// LayerNorm over C cols, bf16 output. grid = rows, block = 256.
__global__ void ln_kernel(const float* __restrict__ x, const float* __restrict__ g,
                          const float* __restrict__ b, bf16* __restrict__ out, int C) {
  __shared__ float red[256];
  int row = blockIdx.x;
  const float* xr = x + (size_t)row * C;
  float s = 0.f;
  for (int c = threadIdx.x; c < C; c += 256) s += xr[c];
  float mu = block_sum(s, red) / (float)C;
  float v = 0.f;
  for (int c = threadIdx.x; c < C; c += 256) { float d = xr[c] - mu; v += d * d; }
  float var = block_sum(v, red) / (float)C;
  float inv = rsqrtf(var + 1e-5f);
  bf16* orow = out + (size_t)row * C;
  for (int c = threadIdx.x; c < C; c += 256)
    orow[c] = (bf16)((xr[c] - mu) * inv * g[c] + b[c]);
}

// RMSNorm (torch-style) over C cols, bf16 output. grid = rows, block = 256.
__global__ void rms_kernel(const float* __restrict__ x, const float* __restrict__ scale,
                           bf16* __restrict__ out, int C) {
  __shared__ float red[256];
  int row = blockIdx.x;
  const float* xr = x + (size_t)row * C;
  float s = 0.f;
  for (int c = threadIdx.x; c < C; c += 256) s += xr[c] * xr[c];
  float ss = block_sum(s, red);
  float rms = sqrtf(ss) * rsqrtf((float)C);
  float inv = 1.0f / (rms + 1e-8f);
  bf16* orow = out + (size_t)row * C;
  for (int c = threadIdx.x; c < C; c += 256)
    orow[c] = (bf16)(xr[c] * inv * scale[c]);
}

// ---------------------------------------------------------------------------
// Generic WMMA GEMM:  C[M x Nc] = epilogue(A[M x K] @ W[K x Nc])
// A bf16 row-major; W packed (pack_w_kernel layout); fp32 accumulation.
// Block = 128 threads (4 waves); block tile 64x64; wave tile 16x64.
// Double-buffered LDS A staging (async global->LDS when available):
// one barrier per K-chunk, stage of chunk k+1 overlaps compute of chunk k.
// Epilogues:
//   0: outB = (bf16)acc                               (qkv)
//   1: outF[row*out_ld + out_off + col] = acc + bias[col] + resid[row*resid_ld+col]
//   2: outB = (bf16)gelu(acc + bias[col])             (fc1)
//   3: split: br=col/256; outF[br*M*256 + row*256 + col%256] += bias/resid   (fc2)
// ---------------------------------------------------------------------------
template <int EP>
__global__ void gemm_wmma(const bf16* __restrict__ A, const unsigned int* __restrict__ Wp,
                          const float* __restrict__ bias, const float* __restrict__ resid,
                          float* __restrict__ outF, bf16* __restrict__ outB,
                          int M, int Nc, int K,
                          int resid_ld, int out_ld, int out_off) {
  __shared__ bf16 sA[2][64 * 32];  // double-buffered A tile, 2 x 4 KB
  const int tid  = threadIdx.x;
  const int lane = tid & 31;
  const int wid  = tid >> 5;
  const int half = lane >> 4;
  const int nl   = lane & 15;
  const int row0 = blockIdx.x * 64;
  const int col0 = blockIdx.y * 64;
  const int tilesN = Nc >> 4;

  v8f acc[4] = {v8f{}, v8f{}, v8f{}, v8f{}};

  // stage one 64x32 bf16 tile: 256 x 16B units, 2 per thread
  auto stage = [&](int k0, int buf) {
#pragma unroll
    for (int it = 0; it < 2; ++it) {
      int idx = tid + it * 128;   // 16B unit index (4 units per 64B row)
      int r = idx >> 2;
      int c = idx & 3;
      async_copy16(A + (size_t)(row0 + r) * K + k0 + c * 8, &sA[buf][r * 32 + c * 8]);
    }
  };

  const int nk = K >> 5;
  stage(0, 0);
  async_wait0();
  __syncthreads();

  for (int kt = 0; kt < nk; ++kt) {
    const int buf = kt & 1;
    if (kt + 1 < nk) stage((kt + 1) * 32, buf ^ 1);

    // A fragment for this wave's 16 rows (ds loads merge into 2x b128)
    Frag16 af;
    const unsigned short* sArow = (const unsigned short*)sA[buf] + (wid * 16 + nl) * 32;
#pragma unroll
    for (int i = 0; i < 8; ++i)
      af.u[i] = *(const unsigned int*)(sArow + a_kbase(i, half));

    // 4 column subtiles; packed W: 8 consecutive dwords per lane -> 2x b128
#pragma unroll
    for (int dt = 0; dt < 4; ++dt) {
      Frag16 bw;
      const unsigned int* wp =
          Wp + (((size_t)kt * tilesN + (col0 >> 4) + dt) * 32 + lane) * 8;
#pragma unroll
      for (int i = 0; i < 8; ++i) bw.u[i] = wp[i];
      acc[dt] = wmma_bf16(af.v, bw.v, acc[dt]);
    }

    if (kt + 1 < nk) {
      async_wait0();
      __syncthreads();
    }
  }

  // epilogue: D layout — VGPR r, lanes 0-15 -> M=r, lanes 16-31 -> M=8+r; N=lane&15
  const int mbase = row0 + wid * 16;
#pragma unroll
  for (int dt = 0; dt < 4; ++dt) {
#pragma unroll
    for (int r = 0; r < 8; ++r) {
      int row = mbase + r + half * 8;
      int col = col0 + dt * 16 + nl;
      float v = acc[dt][r];
      if (EP == 0) {
        outB[(size_t)row * Nc + col] = (bf16)v;
      } else if (EP == 1) {
        outF[(size_t)row * out_ld + out_off + col] =
            v + bias[col] + resid[(size_t)row * resid_ld + col];
      } else if (EP == 2) {
        float t = v + bias[col];
        float gl = 0.5f * t * (1.0f + erff(t * 0.70710678118654752f));
        outB[(size_t)row * Nc + col] = (bf16)gl;
      } else {
        float t = v + bias[col] + resid[(size_t)row * resid_ld + col];
        int br = col >> 8;
        int c  = col & 255;
        outF[(size_t)br * ((size_t)M * 256) + (size_t)row * 256 + c] = t;
      }
    }
  }
}

// ---------------------------------------------------------------------------
// ReLA attention core: out[b,n,h*64+d] = sum_j relu(q.k^T * 1/8)[n,j] * v[j,d]
// qkv: bf16 [B, N, 1536] (q @ col 0, k @ col 512, v @ col 1024, head h at +h*64)
// Block = 128 threads (4 waves); wave -> 16 q rows x DH=64; grid (N/64, B*H).
// Per 32-key chunk: K tile staged row-major in LDS (shared by all 4 waves,
// async copy when available); V tile staged transposed so B-fragments are
// contiguous dwords (2x ds_load_b128 per fragment).
// ---------------------------------------------------------------------------
__global__ void attn_kernel(const bf16* __restrict__ qkv, float* __restrict__ out) {
  const int tid  = threadIdx.x;
  const int bh   = blockIdx.y;
  const int b    = bh / HH;
  const int h    = bh % HH;
  const int lane = tid & 31;
  const int wid  = tid >> 5;
  const int half = lane >> 4;
  const int nl   = lane & 15;
  const int m0   = blockIdx.x * 64 + wid * 16;

  const unsigned short* qk = (const unsigned short*)qkv;
  const size_t base = (size_t)b * NN * 1536;
  const int qoff = h * DH;
  const int koff = INNER + h * DH;
  const int voff = 2 * INNER + h * DH;

  __shared__ bf16 sS[4][16 * 32];   // per-wave 16x32 score tile (D->A relayout)
  __shared__ bf16 sK[32 * 64];      // K chunk row-major: [key][dh]
  __shared__ bf16 sVt[64 * 32];     // V chunk transposed: [dh][key]
  bf16* myS = sS[wid];

  // Q A-fragments: row m0+nl, K = 0..31 and 32..63
  Frag16 aq[2];
  {
    const unsigned short* qrow = qk + base + (size_t)(m0 + nl) * 1536 + qoff;
#pragma unroll
    for (int c = 0; c < 2; ++c)
#pragma unroll
      for (int i = 0; i < 8; ++i)
        aq[c].u[i] = *(const unsigned int*)(qrow + c * 32 + a_kbase(i, half));
  }

  v8f acc[4] = {v8f{}, v8f{}, v8f{}, v8f{}};
  const float scale = 0.125f;  // DH^-0.5

  for (int j0 = 0; j0 < NN; j0 += 32) {
    // --- cooperative K staging (row-major, async): 256 x 16B units, 2/thread
#pragma unroll
    for (int it = 0; it < 2; ++it) {
      int idx = tid + it * 128;  // 8 units per 128B key row
      int r = idx >> 3;
      int c = idx & 7;
      async_copy16(qk + base + (size_t)(j0 + r) * 1536 + koff + c * 8,
                   (bf16*)sK + r * 64 + c * 8);
    }

    // --- cooperative V staging (transposed): thread -> (key, 16-dh segment)
    {
      int key = tid >> 2;            // 0..31
      int dh0 = (tid & 3) * 16;      // 0,16,32,48
      const unsigned short* vrow = qk + base + (size_t)(j0 + key) * 1536 + voff + dh0;
      unsigned short tmp[16];
      *(uint4*)(tmp)     = *(const uint4*)(vrow);
      *(uint4*)(tmp + 8) = *(const uint4*)(vrow + 8);
      unsigned short* sv = (unsigned short*)sVt;
#pragma unroll
      for (int jj = 0; jj < 16; ++jj)
        sv[(dh0 + jj) * 32 + key] = tmp[jj];
    }

    async_wait0();
    __syncthreads();  // sK / sVt staged

    // --- scores for two 16-key subtiles; B = K^T so lane nl reads key row jt*16+nl
#pragma unroll
    for (int jt = 0; jt < 2; ++jt) {
      const unsigned short* krow = (const unsigned short*)sK + (jt * 16 + nl) * 64;
      Frag16 bk0, bk1;
#pragma unroll
      for (int i = 0; i < 8; ++i) {
        int kk = half * 16 + 2 * i;            // contiguous pair within the key row
        bk0.u[i] = *(const unsigned int*)(krow + kk);
        bk1.u[i] = *(const unsigned int*)(krow + 32 + kk);
      }
      v8f s = v8f{};
      s = wmma_bf16(aq[0].v, bk0.v, s);
      s = wmma_bf16(aq[1].v, bk1.v, s);
      // relu(scale * s) -> LDS as bf16 (D layout: M = r + 8*half, N = nl)
#pragma unroll
      for (int r = 0; r < 8; ++r) {
        float val = s[r] * scale;
        val = val > 0.f ? val : 0.f;
        myS[(r + half * 8) * 32 + jt * 16 + nl] = (bf16)val;
      }
    }

    // --- reload score tile as A fragment (same-wave LDS is in-order)
    Frag16 as;
    const unsigned short* srow = (const unsigned short*)myS + nl * 32;
#pragma unroll
    for (int i = 0; i < 8; ++i)
      as.u[i] = *(const unsigned int*)(srow + a_kbase(i, half));

    // --- V fragments from transposed LDS: 8 consecutive dwords per lane
    const unsigned short* sv = (const unsigned short*)sVt;
#pragma unroll
    for (int dt = 0; dt < 4; ++dt) {
      Frag16 bv;
      const unsigned short* svrow = sv + (dt * 16 + nl) * 32 + half * 16;
#pragma unroll
      for (int i = 0; i < 8; ++i)
        bv.u[i] = *(const unsigned int*)(svrow + 2 * i);
      acc[dt] = wmma_bf16(as.v, bv.v, acc[dt]);
    }

    __syncthreads();  // all consumers done before next stage overwrites
  }

  // write out [B, N, 512]
  float* orow = out + (size_t)b * NN * INNER;
#pragma unroll
  for (int dt = 0; dt < 4; ++dt)
#pragma unroll
    for (int r = 0; r < 8; ++r) {
      int row = m0 + r + half * 8;
      orow[(size_t)row * INNER + h * DH + dt * 16 + nl] = acc[dt][r];
    }
}

// ---------------------------------------------------------------------------
// Host-side orchestration
// ---------------------------------------------------------------------------
static inline size_t align256(size_t x) { return (x + 255) & ~(size_t)255; }

extern "C" void kernel_launch(void* const* d_in, const int* in_sizes, int n_in,
                              void* d_out, int out_size, void* d_ws, size_t ws_size,
                              hipStream_t stream) {
  (void)in_sizes; (void)n_in; (void)out_size; (void)ws_size;

  // inputs
  const float* x[3]  = {(const float*)d_in[0], (const float*)d_in[1], (const float*)d_in[2]};
  const float *g1[3], *b1[3], *wqkv[3], *rmsw[3], *wo[3], *bo[3];
  for (int br = 0; br < 3; ++br) {
    int base = 3 + 6 * br;
    g1[br]   = (const float*)d_in[base + 0];
    b1[br]   = (const float*)d_in[base + 1];
    wqkv[br] = (const float*)d_in[base + 2];
    rmsw[br] = (const float*)d_in[base + 3];
    wo[br]   = (const float*)d_in[base + 4];
    bo[br]   = (const float*)d_in[base + 5];
  }
  const float* g2    = (const float*)d_in[21];
  const float* b2    = (const float*)d_in[22];
  const float* w_fc1 = (const float*)d_in[23];
  const float* b_fc1 = (const float*)d_in[24];
  const float* w_fc2 = (const float*)d_in[25];
  const float* b_fc2 = (const float*)d_in[26];
  float* outp = (float*)d_out;

  // workspace carve-up
  char* ws = (char*)d_ws;
  size_t off = 0;
  auto carve = [&](size_t bytes) { char* p = ws + off; off += align256(bytes); return p; };
  unsigned int* wqkv_p[3]; unsigned int* wo_p[3];
  for (int br = 0; br < 3; ++br) wqkv_p[br] = (unsigned int*)carve((size_t)DIM * 3 * INNER * 2);
  for (int br = 0; br < 3; ++br) wo_p[br]   = (unsigned int*)carve((size_t)INNER * DIM * 2);
  unsigned int* wfc1_p = (unsigned int*)carve((size_t)D3 * HID * 2);
  unsigned int* wfc2_p = (unsigned int*)carve((size_t)HID * D3 * 2);
  bf16*  xln    = (bf16*)carve((size_t)ROWS * DIM * 2);
  bf16*  qkvb   = (bf16*)carve((size_t)ROWS * 3 * INNER * 2);
  float* attno  = (float*)carve((size_t)ROWS * INNER * 4);
  bf16*  rmsb   = (bf16*)carve((size_t)ROWS * INNER * 2);
  float* xcat   = (float*)carve((size_t)ROWS * D3 * 4);
  bf16*  xln2   = (bf16*)carve((size_t)ROWS * D3 * 2);
  bf16*  hbuf   = (bf16*)carve((size_t)ROWS * HID * 2);

  auto pack = [&](const float* src, unsigned int* dst, int K, int Nc) {
    int total = (K * Nc) >> 1;
    pack_w_kernel<<<(total + 255) / 256, 256, 0, stream>>>(src, dst, K, Nc);
  };
  for (int br = 0; br < 3; ++br) pack(wqkv[br], wqkv_p[br], DIM, 3 * INNER);
  for (int br = 0; br < 3; ++br) pack(wo[br], wo_p[br], INNER, DIM);
  pack(w_fc1, wfc1_p, D3, HID);
  pack(w_fc2, wfc2_p, HID, D3);

  for (int br = 0; br < 3; ++br) {
    // LN1 -> bf16
    ln_kernel<<<ROWS, 256, 0, stream>>>(x[br], g1[br], b1[br], xln, DIM);
    // qkv GEMM: [4096,256] @ [256,1536] -> bf16
    gemm_wmma<0><<<dim3(ROWS / 64, (3 * INNER) / 64), 128, 0, stream>>>(
        xln, wqkv_p[br], nullptr, nullptr, nullptr, qkvb,
        ROWS, 3 * INNER, DIM, 0, 0, 0);
    // ReLA attention
    attn_kernel<<<dim3(NN / 64, BB * HH), 128, 0, stream>>>(qkvb, attno);
    // RMSNorm -> bf16
    rms_kernel<<<ROWS, 256, 0, stream>>>(attno, rmsw[br], rmsb, INNER);
    // wo GEMM + bias + residual -> xcat column block
    gemm_wmma<1><<<dim3(ROWS / 64, DIM / 64), 128, 0, stream>>>(
        rmsb, wo_p[br], bo[br], x[br], xcat, nullptr,
        ROWS, DIM, INNER, DIM, D3, br * DIM);
  }

  // LN2 over concat
  ln_kernel<<<ROWS, 256, 0, stream>>>(xcat, g2, b2, xln2, D3);
  // fc1 + gelu -> bf16
  gemm_wmma<2><<<dim3(ROWS / 64, HID / 64), 128, 0, stream>>>(
      xln2, wfc1_p, b_fc1, nullptr, nullptr, hbuf,
      ROWS, HID, D3, 0, 0, 0);
  // fc2 + bias + residual(xcat) -> split fp32 outputs
  gemm_wmma<3><<<dim3(ROWS / 64, D3 / 64), 128, 0, stream>>>(
      hbuf, wfc2_p, b_fc2, xcat, outp, nullptr,
      ROWS, D3, HID, D3, 0, 0);
}
